// SphericalChebConv_36361193128503
// MI455X (gfx1250) — matrix operand order repaired and checked by reading, not compiled, not run on hardware
//
#include <hip/hip_runtime.h>
#include <hip/hip_bf16.h>

// ---------------------------------------------------------------------------
// SphericalChebConv for MI455X (gfx1250, wave32, WMMA + TDM)
//   N=10000 nodes, CIN=COUT=512, K_cheb=5, knn=20
//   * constant edge weight -1/knn (out-degree is exactly knn for every node)
//   * prop() = gather over transpose adjacency (reverse CSR built once)
//   * 5 GEMMs [10000x512]@[512x512] -> v_wmma_f32_16x16x32_bf16,
//     A tile staged in LDS by the Tensor Data Mover (tensor_load_to_lds)
// ---------------------------------------------------------------------------

#define NPTS  10000
#define CIN   512
#define COUT  512
#define KCH   5
#define KNN   20
#define NE    (NPTS * KNN)

// LDS A-tile row stride in dwords: 512 + 4-dword pad every 256 dwords
#define AROW  520

typedef float    v8f   __attribute__((ext_vector_type(8)));
typedef float    v16f  __attribute__((ext_vector_type(16)));
typedef __bf16   v16bf __attribute__((ext_vector_type(16)));
typedef unsigned u32x4 __attribute__((ext_vector_type(4)));
typedef int      i32x4 __attribute__((ext_vector_type(4)));
typedef int      i32x8 __attribute__((ext_vector_type(8)));

#if __has_builtin(__builtin_amdgcn_tensor_load_to_lds)
#define HAVE_TDM 1
#else
#define HAVE_TDM 0
#endif

__device__ __host__ inline unsigned short f2bf_bits(float f) {
    union { float f; unsigned u; } t; t.f = f;
    unsigned u = t.u;
    unsigned r = u + 0x7FFFu + ((u >> 16) & 1u);   // round-to-nearest-even
    return (unsigned short)(r >> 16);
}

// ---------------------------------------------------------------------------
// 1) transpose x [CIN, N] -> h [N, CIN]
// ---------------------------------------------------------------------------
__global__ __launch_bounds__(256) void k_transpose_x(const float* __restrict__ x,
                                                     float* __restrict__ h) {
    __shared__ float tile[32][33];
    int n0 = blockIdx.x * 32, c0 = blockIdx.y * 32;
    int tx = threadIdx.x & 31, ty = threadIdx.x >> 5;
    for (int r = ty; r < 32; r += 8) {
        int n = n0 + tx;
        tile[r][tx] = (n < NPTS) ? x[(c0 + r) * NPTS + n] : 0.0f;
    }
    __syncthreads();
    for (int r = ty; r < 32; r += 8) {
        int n = n0 + r;
        if (n < NPTS) h[n * CIN + (c0 + tx)] = tile[tx][r];
    }
}

// ---------------------------------------------------------------------------
// 2) theta [K][CIN][COUT] f32 -> thetaT [K][COUT][CIN] bf16
// ---------------------------------------------------------------------------
__global__ __launch_bounds__(256) void k_theta_bf16(const float* __restrict__ theta,
                                                    __bf16* __restrict__ thT) {
    __shared__ float tile[32][33];
    int k  = blockIdx.z;
    int c0 = blockIdx.x * 32;
    int o0 = blockIdx.y * 32;
    const float* th = theta + (size_t)k * CIN * COUT;
    __bf16* dst = thT + (size_t)k * CIN * COUT;
    int tx = threadIdx.x & 31, ty = threadIdx.x >> 5;
    for (int r = ty; r < 32; r += 8)
        tile[r][tx] = th[(c0 + r) * COUT + (o0 + tx)];
    __syncthreads();
    for (int r = ty; r < 32; r += 8) {
        union { unsigned short s; __bf16 b; } u;
        u.s = f2bf_bits(tile[tx][r]);
        dst[(o0 + r) * CIN + (c0 + tx)] = u.b;   // dst[o][c] = theta[c][o]
    }
}

// ---------------------------------------------------------------------------
// 3) brute-force kNN: one wave32 per point; per-lane sorted top-20 in LDS,
//    then a 32-way shuffle merge.
// ---------------------------------------------------------------------------
__global__ __launch_bounds__(256) void k_knn(const float* __restrict__ pos,
                                             int* __restrict__ nbr) {
    __shared__ float sd[8 * KNN * 32];
    __shared__ int   si[8 * KNN * 32];
    int wave = threadIdx.x >> 5, lane = threadIdx.x & 31;
    int i = blockIdx.x * 8 + wave;                       // 1250*8 == NPTS
    float px = pos[3 * i], py = pos[3 * i + 1], pz = pos[3 * i + 2];

    float* d  = &sd[(wave * KNN) * 32 + lane];
    int*   id = &si[(wave * KNN) * 32 + lane];
    for (int r = 0; r < KNN; ++r) { d[r * 32] = 3.0e38f; id[r * 32] = -1; }

    for (int c = lane; c < NPTS; c += 32) {
        if (c == i) continue;
        float dx = pos[3 * c] - px, dy = pos[3 * c + 1] - py, dz = pos[3 * c + 2] - pz;
        float d2 = dx * dx + dy * dy + dz * dz;
        if (d2 < d[(KNN - 1) * 32]) {
            int p = KNN - 1;
            while (p > 0 && d[(p - 1) * 32] > d2) {
                d[p * 32] = d[(p - 1) * 32]; id[p * 32] = id[(p - 1) * 32]; --p;
            }
            d[p * 32] = d2; id[p * 32] = c;
        }
    }
    int ptr = 0;
    for (int r = 0; r < KNN; ++r) {
        float md = (ptr < KNN) ? d[ptr * 32] : 3.0e38f;
        int   mi = (ptr < KNN) ? id[ptr * 32] : -1;
        int   ml = lane;
        for (int off = 16; off; off >>= 1) {
            float od = __shfl_xor(md, off, 32);
            int   oi = __shfl_xor(mi, off, 32);
            int   ol = __shfl_xor(ml, off, 32);
            if (od < md || (od == md && oi < mi)) { md = od; mi = oi; ml = ol; }
        }
        if (ml == lane) ++ptr;
        if (lane == 0) nbr[i * KNN + r] = mi;
    }
}

// ---------------------------------------------------------------------------
// 4) reverse-CSR build
// ---------------------------------------------------------------------------
__global__ __launch_bounds__(256) void k_zero(int* __restrict__ cnt) {
    int i = blockIdx.x * 256 + threadIdx.x;
    if (i <= NPTS) cnt[i] = 0;
}
__global__ __launch_bounds__(256) void k_count(const int* __restrict__ nbr,
                                               int* __restrict__ cnt) {
    int e = blockIdx.x * 256 + threadIdx.x;
    if (e < NE) atomicAdd(&cnt[nbr[e]], 1);
}
__global__ __launch_bounds__(256) void k_scan(int* __restrict__ cnt,
                                              int* __restrict__ offs) {
    __shared__ int buf[256];
    __shared__ int run_s;
    int tid = threadIdx.x;
    if (tid == 0) run_s = 0;
    __syncthreads();
    for (int base = 0; base < NPTS; base += 256) {
        int i = base + tid;
        int v = (i < NPTS) ? cnt[i] : 0;
        buf[tid] = v;
        __syncthreads();
        for (int off = 1; off < 256; off <<= 1) {
            int a = (tid >= off) ? buf[tid - off] : 0;
            __syncthreads();
            buf[tid] += a;
            __syncthreads();
        }
        int incl = buf[tid];
        int base_run = run_s;
        if (i < NPTS) { offs[i] = base_run + incl - v; cnt[i] = 0; }
        __syncthreads();
        if (tid == 0) run_s += buf[255];
        __syncthreads();
    }
    if (tid == 0) offs[NPTS] = run_s;
}
__global__ __launch_bounds__(256) void k_fill(const int* __restrict__ nbr,
                                              const int* __restrict__ offs,
                                              int* __restrict__ cursor,
                                              int* __restrict__ srcs) {
    int e = blockIdx.x * 256 + threadIdx.x;
    if (e >= NE) return;
    int i = e / KNN;
    int j = nbr[e];
    int p = atomicAdd(&cursor[j], 1);
    srcs[offs[j] + p] = i;
}

// ---------------------------------------------------------------------------
// 5) prop: dst[j,:] = w2 * sum_{i in rev[j]} src[i,:]  (- prev[j,:] if dosub)
// ---------------------------------------------------------------------------
__global__ __launch_bounds__(256) void k_prop(const float* __restrict__ src,
                                              const float* __restrict__ prev,
                                              const int* __restrict__ offs,
                                              const int* __restrict__ srcs,
                                              float* __restrict__ dst,
                                              float w2, int dosub) {
    int j = blockIdx.x;
    int beg = offs[j], end = offs[j + 1];
    int c0 = threadIdx.x, c1 = threadIdx.x + 256;
    float s0 = 0.0f, s1 = 0.0f;
    for (int e = beg; e < end; ++e) {
        const float* tp = src + (size_t)srcs[e] * CIN;
        s0 += tp[c0];
        s1 += tp[c1];
    }
    float r0 = w2 * s0, r1 = w2 * s1;
    if (dosub) {
        r0 -= prev[(size_t)j * CIN + c0];
        r1 -= prev[(size_t)j * CIN + c1];
    }
    dst[(size_t)j * CIN + c0] = r0;
    dst[(size_t)j * CIN + c1] = r1;
}

// ---------------------------------------------------------------------------
// 6) WMMA GEMM: acc[N,COUT] (+)= A[N,CIN](f32) @ theta  (Bt = [COUT,CIN] bf16)
//    - A tile (16 rows x 512 K, f32) staged in LDS once per block by the TDM
//      (padding: +4 dwords every 256 dwords -> row stride 520 dwords)
//    - each of the 8 waves owns a 16x64 output strip
//    - K-loop unrolled x2 so one iteration's B loads overlap the other's WMMAs
//    - 625 blocks x 16 rows == 10000 exactly -> EXEC all-ones, no guards
// ---------------------------------------------------------------------------
__device__ inline v16bf pack_bf16(float4 a, float4 b, float4 c, float4 d) {
    v16f vf = { a.x, a.y, a.z, a.w,
                b.x, b.y, b.z, b.w,
                c.x, c.y, c.z, c.w,
                d.x, d.y, d.z, d.w };
    return __builtin_convertvector(vf, v16bf);   // native v_cvt_pk_bf16_f32
}

template <int INIT>
__global__ __launch_bounds__(256) void k_gemm_wmma(const float* __restrict__ A,
                                                   const __bf16* __restrict__ Bt,
                                                   float* __restrict__ acc) {
    __shared__ float atile[16 * AROW + 16];   // ~33.3 KB (incl. trailing pad)

    const int lane = threadIdx.x & 31;
    const int wave = threadIdx.x >> 5;     // channel group: cols wave*64 .. +63
    const int tm   = blockIdx.x;           // node tile: rows tm*16 .. +15
    const int l16  = lane & 15;
    const int hi   = lane >> 4;

#if HAVE_TDM
    if (wave == 0) {
        // --- Tensor DMA: 2D tile (512 x 16) of f32 from A into LDS ---
        unsigned lds_off = (unsigned)(unsigned long long)(atile);  // LDS aperture: low 32 bits
        unsigned long long ga =
            (unsigned long long)(const void*)(A + (size_t)tm * 16 * CIN);
        // D# group 0: count=1 | lds_addr | global_addr | type=2
        u32x4 g0 = { 1u, lds_off, (unsigned)ga,
                     (unsigned)((ga >> 32) & 0x1FFFFFFull) | 0x80000000u };
        // D# group 1: data_size=4B, pad_enable, pad_interval=256dw, pad=4dw,
        //             tensor_dim0=512, tensor_dim1=16, tile_dim0=512, tile_dim1=16,
        //             tensor_dim0_stride=512
        i32x8 g1 = { (int)((2u << 16) | (1u << 20) | (7u << 22) | (3u << 25)),
                     (int)(512u << 16),          // tensor_dim0[15:0]
                     (int)(16u << 16),           // tensor_dim1[15:0]
                     (int)(512u << 16),          // tile_dim0
                     16,                         // tile_dim1
                     512,                        // tensor_dim0_stride[31:0]
                     0, 0 };
        i32x4 gz = { 0, 0, 0, 0 };
#if __clang_major__ >= 23
        i32x8 gz8 = { 0, 0, 0, 0, 0, 0, 0, 0 };
        __builtin_amdgcn_tensor_load_to_lds(g0, g1, gz, gz, gz8, 0);
#else
        __builtin_amdgcn_tensor_load_to_lds(g0, g1, gz, gz, 0);
#endif
        __builtin_amdgcn_s_wait_tensorcnt(0);
    }
#else
    {
        // cooperative fallback: 16 rows x 512 cols, same padded layout
        int row = threadIdx.x >> 4;
        int cb  = (threadIdx.x & 15) * 32;
        const float* src = A + (size_t)(tm * 16 + row) * CIN + cb;
        float* dstr = &atile[row * AROW + cb + 4 * (cb >> 8)];
        for (int c = 0; c < 32; c += 4)
            *(float4*)(dstr + c) = *(const float4*)(src + c);
    }
#endif
    __syncthreads();

    const float*  arow = &atile[l16 * AROW];
    const __bf16* b0   = Bt + (size_t)(wave * 64 + l16) * CIN + hi * 16;

    v8f z = {0.f, 0.f, 0.f, 0.f, 0.f, 0.f, 0.f, 0.f};
    v8f c0 = z, c1 = z, c2 = z, c3 = z;

#pragma unroll 2
    for (int kk = 0; kk < CIN; kk += 32) {
        // B fragments (32x16 bf16): lane<16 -> K 0..15 of col N=l16,
        //                           lane>=16 -> K 16..31 (contiguous 32B loads)
        v16bf bf0 = *(const v16bf*)(b0 + 0 * 16 * CIN + kk);
        v16bf bf1 = *(const v16bf*)(b0 + 1 * 16 * CIN + kk);
        v16bf bf2 = *(const v16bf*)(b0 + 2 * 16 * CIN + kk);
        v16bf bf3 = *(const v16bf*)(b0 + 3 * 16 * CIN + kk);

        // A fragment from LDS (padded row layout, 16B aligned)
        int kp = kk + 4 * (kk >> 8);
        float4 a0 = *(const float4*)(arow + kp + hi * 16);
        float4 a1 = *(const float4*)(arow + kp + hi * 16 + 4);
        float4 a2 = *(const float4*)(arow + kp + hi * 16 + 8);
        float4 a3 = *(const float4*)(arow + kp + hi * 16 + 12);
        v16bf af = pack_bf16(a0, a1, a2, a3);

        c0 = __builtin_amdgcn_wmma_f32_16x16x32_bf16(false, af, false, bf0,
                                                     (short)0, c0, false, false);
        c1 = __builtin_amdgcn_wmma_f32_16x16x32_bf16(false, af, false, bf1,
                                                     (short)0, c1, false, false);
        c2 = __builtin_amdgcn_wmma_f32_16x16x32_bf16(false, af, false, bf2,
                                                     (short)0, c2, false, false);
        c3 = __builtin_amdgcn_wmma_f32_16x16x32_bf16(false, af, false, bf3,
                                                     (short)0, c3, false, false);
    }

    // C/D layout: vgpr r -> lanes 0-15: (M=r, N=lane); lanes 16-31: (M=8+r)
    float* outp = acc + (size_t)(tm * 16 + hi * 8) * COUT + wave * 64 + l16;
    v8f cs[4] = {c0, c1, c2, c3};
#pragma unroll
    for (int t = 0; t < 4; ++t) {
#pragma unroll
        for (int r = 0; r < 8; ++r) {
            float* p = outp + (size_t)r * COUT + t * 16;
            if (INIT) *p = cs[t][r];
            else      *p += cs[t][r];
        }
    }
}

// ---------------------------------------------------------------------------
// 7) out[c, n] = acc[n, c] + bias[c]
// ---------------------------------------------------------------------------
__global__ __launch_bounds__(256) void k_out_bias(const float* __restrict__ acc,
                                                  const float* __restrict__ bias,
                                                  float* __restrict__ out) {
    __shared__ float tile[32][33];
    int n0 = blockIdx.x * 32, c0 = blockIdx.y * 32;
    int tx = threadIdx.x & 31, ty = threadIdx.x >> 5;
    for (int r = ty; r < 32; r += 8) {
        int n = n0 + r;
        tile[r][tx] = (n < NPTS) ? acc[(size_t)n * COUT + c0 + tx] : 0.0f;
    }
    __syncthreads();
    for (int r = ty; r < 32; r += 8) {
        int n = n0 + tx;
        if (n < NPTS) out[(size_t)(c0 + r) * NPTS + n] = tile[tx][r] + bias[c0 + r];
    }
}

// ---------------------------------------------------------------------------
// host-side launch
// ---------------------------------------------------------------------------
extern "C" void kernel_launch(void* const* d_in, const int* in_sizes, int n_in,
                              void* d_out, int out_size, void* d_ws, size_t ws_size,
                              hipStream_t stream) {
    (void)in_sizes; (void)n_in; (void)out_size; (void)ws_size;
    const float* x        = (const float*)d_in[0];   // [1, CIN, N]
    const float* position = (const float*)d_in[1];   // [1, N, 3]
    const float* theta    = (const float*)d_in[2];   // [K, CIN, COUT]
    const float* bias     = (const float*)d_in[3];   // [COUT]
    float* out = (float*)d_out;                       // [1, COUT, N]

    char* ws = (char*)d_ws;
    size_t off = 0;
    auto carve = [&](size_t bytes) {
        void* p = ws + off;
        off += (bytes + 255) & ~(size_t)255;
        return p;
    };
    int*    nbr  = (int*)carve((size_t)NE * 4);
    int*    cnt  = (int*)carve((size_t)(NPTS + 1) * 4);
    int*    offs = (int*)carve((size_t)(NPTS + 1) * 4);
    int*    srcs = (int*)carve((size_t)NE * 4);
    float*  TxA  = (float*)carve((size_t)NPTS * CIN * 4);
    float*  TxB  = (float*)carve((size_t)NPTS * CIN * 4);
    float*  TxC  = (float*)carve((size_t)NPTS * CIN * 4);
    float*  acc  = (float*)carve((size_t)NPTS * COUT * 4);
    __bf16* thT  = (__bf16*)carve((size_t)KCH * CIN * COUT * 2 + 64);

    const float W = -1.0f / (float)KNN;

    k_transpose_x<<<dim3((NPTS + 31) / 32, CIN / 32), 256, 0, stream>>>(x, TxA);
    k_theta_bf16<<<dim3(CIN / 32, COUT / 32, KCH), 256, 0, stream>>>(theta, thT);

    k_knn<<<NPTS / 8, 256, 0, stream>>>(position, nbr);
    k_zero<<<(NPTS + 256) / 256, 256, 0, stream>>>(cnt);
    k_count<<<(NE + 255) / 256, 256, 0, stream>>>(nbr, cnt);
    k_scan<<<1, 256, 0, stream>>>(cnt, offs);
    k_fill<<<(NE + 255) / 256, 256, 0, stream>>>(nbr, offs, cnt, srcs);

    const size_t TH = (size_t)CIN * COUT;

    k_gemm_wmma<1><<<NPTS / 16, 256, 0, stream>>>(TxA, thT + 0 * TH, acc);
    k_prop<<<NPTS, 256, 0, stream>>>(TxA, TxA, offs, srcs, TxB, W, 0);          // Tx1
    k_gemm_wmma<0><<<NPTS / 16, 256, 0, stream>>>(TxB, thT + 1 * TH, acc);
    k_prop<<<NPTS, 256, 0, stream>>>(TxB, TxA, offs, srcs, TxC, 2.0f * W, 1);   // Tx2
    k_gemm_wmma<0><<<NPTS / 16, 256, 0, stream>>>(TxC, thT + 2 * TH, acc);
    k_prop<<<NPTS, 256, 0, stream>>>(TxC, TxB, offs, srcs, TxA, 2.0f * W, 1);   // Tx3
    k_gemm_wmma<0><<<NPTS / 16, 256, 0, stream>>>(TxA, thT + 3 * TH, acc);
    k_prop<<<NPTS, 256, 0, stream>>>(TxA, TxC, offs, srcs, TxB, 2.0f * W, 1);   // Tx4
    k_gemm_wmma<0><<<NPTS / 16, 256, 0, stream>>>(TxB, thT + 4 * TH, acc);

    k_out_bias<<<dim3((NPTS + 31) / 32, COUT / 32), 256, 0, stream>>>(acc, bias, out);
}
